// RQVAE_4140348473636
// MI455X (gfx1250) — matrix operand order repaired and checked by reading, not compiled, not run on hardware
//
#include <hip/hip_runtime.h>
#include <hip/hip_bf16.h>

// ---------------------------------------------------------------------------
// RQ-VAE forward on gfx1250: 8 WMMA GEMMs + residual VQ stage.
// B=131072, encoder 768->512->256->128->32, 3 RQ layers (256x32 codebooks),
// decoder 32->128->256->512->768.
// ---------------------------------------------------------------------------

#define RQB        131072
#define D_IN       768
#define D_LAT      32
#define N_EMBED    256
#define N_LAYERS   3

#define XHAT_ELEMS ((size_t)RQB * D_IN)          // 100663296
#define LOSS_OFF   XHAT_ELEMS                    // codebook_loss
#define COMMIT_OFF (XHAT_ELEMS + 1)              // commitment_loss
#define IDX_OFF    (XHAT_ELEMS + 2)              // indices (B,3) as float
#define LOSS_SCALE (1.0f / ((float)RQB * (float)D_LAT))

typedef __attribute__((ext_vector_type(16))) _Float16 v16h;
typedef __attribute__((ext_vector_type(8)))  _Float16 v8h;
typedef __attribute__((ext_vector_type(8)))  float    v8f;
typedef __attribute__((ext_vector_type(4)))  float    v4f;

union Frag16 {
    v16h     v;
    v8h      h8[2];
    _Float16 h[16];
};

// ---------------------------------------------------------------------------
// Weight conversion: W (KxN, f32 row-major) -> WMMA-B-fragment-major f16:
//   Wf[(((kb*N)+n)*2+g)*16 + e] = W[kb*32 + g*16 + e][n]
// so column n's 32 K values are 64 contiguous bytes; lane half g reads K 8g..8g+7 and 16+8g..16+8g+7.
// ---------------------------------------------------------------------------
__global__ __launch_bounds__(256) void convert_w_kernel(
    const float* __restrict__ w, _Float16* __restrict__ wf, int K, int N) {
    int id = blockIdx.x * 256 + threadIdx.x;
    if (id >= K * N) return;
    int k = id / N, n = id % N;
    int kb = k >> 5, g = (k >> 4) & 1, e = k & 15;
    wf[(((size_t)kb * N + n) * 2 + g) * 16 + e] = (_Float16)w[id];
}

__device__ __forceinline__ float silu_f(float v) {
    return v * (1.0f / (1.0f + __expf(-v)));
}

// ---------------------------------------------------------------------------
// Tiled GEMM: C[B x N] = act( A[B x K] @ W[K x N] )
// 256 threads = 8 waves; each wave owns a 16-row stripe of a 128-row block
// and iterates N in compile-time chunks of NCHUNK 16-wide tiles, K in steps
// of 32. A (16x32 f16) / B (32x16 f16) fragments follow the CDNA5 ISA lane
// layouts; accumulate with v_wmma_f32_16x16x32_f16. NCHUNK must divide N/16
// so the inner loop is branch-free straight-line WMMA code.
// ---------------------------------------------------------------------------
template <bool SILU, bool IN_F32, bool OUT_F32, int NCHUNK>
__global__ __launch_bounds__(256) void gemm_wmma_kernel(
    const void* __restrict__ Ain, const _Float16* __restrict__ Wf,
    void* __restrict__ Cout, int K, int N) {
    const int lane    = threadIdx.x & 31;
    const int wave    = threadIdx.x >> 5;
    const int rowBase = blockIdx.x * 128 + wave * 16;
    const int mrow    = rowBase + (lane & 15);
    const int g       = lane >> 4;      // half-wave K group
    const int ncol    = lane & 15;      // column within 16-wide tile

    const _Float16* Ah = (const _Float16*)Ain;
    const float*    Af = (const float*)Ain;

    const int nTiles  = N >> 4;
    const int kBlocks = K >> 5;

    for (int nt0 = 0; nt0 < nTiles; nt0 += NCHUNK) {
        v8f acc[NCHUNK] = {};

        for (int kb = 0; kb < kBlocks; ++kb) {
            // ---- A fragment: lane g covers K = kb*32 + g*8 + [0..7] and +16
            Frag16 a;
            const int k0 = kb * 32 + g * 8;
            if (IN_F32) {
                const float* p = Af + (size_t)mrow * K + k0;
                v4f f0 = *(const v4f*)(p);
                v4f f1 = *(const v4f*)(p + 4);
                v4f f2 = *(const v4f*)(p + 16);
                v4f f3 = *(const v4f*)(p + 20);
#pragma unroll
                for (int e = 0; e < 4; ++e) {
                    a.h[e]      = (_Float16)f0[e];
                    a.h[4 + e]  = (_Float16)f1[e];
                    a.h[8 + e]  = (_Float16)f2[e];
                    a.h[12 + e] = (_Float16)f3[e];
                }
            } else {
                const _Float16* p = Ah + (size_t)mrow * K + k0;
                a.h8[0] = *(const v8h*)(p);
                a.h8[1] = *(const v8h*)(p + 16);
            }

            // ---- NCHUNK independent B fragments + WMMAs (straight-line)
            const _Float16* bbase =
                Wf + (((size_t)kb * N + nt0 * 16 + ncol) * 2 + g) * 16;
#pragma unroll
            for (int j = 0; j < NCHUNK; ++j) {
                const _Float16* bp = bbase + (size_t)j * 16 * 32;  // 16 cols
                Frag16 b;
                b.h8[0] = *(const v8h*)(bp - g * 8);
                b.h8[1] = *(const v8h*)(bp + 16 - g * 8);
                acc[j] = __builtin_amdgcn_wmma_f32_16x16x32_f16(
                    false, a.v, false, b.v, (short)0, acc[j], false, false);
            }
        }

        // ---- epilogue: D VGPR r -> row rowBase + g*8 + r, col n
#pragma unroll
        for (int j = 0; j < NCHUNK; ++j) {
            const int n = (nt0 + j) * 16 + ncol;
#pragma unroll
            for (int r = 0; r < 8; ++r) {
                float v = acc[j][r];
                if (SILU) v = silu_f(v);
                const size_t row = (size_t)rowBase + g * 8 + r;
                if (OUT_F32)
                    ((float*)Cout)[row * N + n] = v;
                else
                    ((_Float16*)Cout)[row * N + n] = (_Float16)v;
            }
        }
    }
}

// ---------------------------------------------------------------------------
// Residual VQ: per-row normalize -> argmin over 256 codes (x3 layers).
// Codebook layer staged in LDS with precomputed ||cb||^2. Forward value of the
// straight-through path: z_q_total += cb[idx]; residual = x_in - cb[idx];
// codebook_loss == commitment_loss == sum mean((cb[idx]-x_in)^2).
// ---------------------------------------------------------------------------
__global__ __launch_bounds__(256) void vq_kernel(
    const float* __restrict__ z, const float* __restrict__ codebooks,
    _Float16* __restrict__ zq_out, float* __restrict__ out,
    float* __restrict__ partial) {
    __shared__ float cb[N_EMBED * D_LAT];   // 32 KB
    __shared__ float cbn[N_EMBED];
    __shared__ float red[256];

    const int    t   = threadIdx.x;
    const size_t row = (size_t)blockIdx.x * 256 + t;

    float res[D_LAT], zqt[D_LAT];
    const float* zr = z + row * D_LAT;
#pragma unroll
    for (int i = 0; i < D_LAT; ++i) { res[i] = zr[i]; zqt[i] = 0.0f; }

    float lsum = 0.0f;
    int   idx[N_LAYERS];

    for (int l = 0; l < N_LAYERS; ++l) {
        __syncthreads();
        // thread t stages codebook row t + its squared norm
        const float* cl = codebooks + ((size_t)l * N_EMBED) * D_LAT + t * D_LAT;
        float nrm = 0.0f;
#pragma unroll
        for (int i = 0; i < D_LAT; i += 4) {
            v4f v = *(const v4f*)(cl + i);
            *(v4f*)&cb[t * D_LAT + i] = v;
            nrm += v[0] * v[0] + v[1] * v[1] + v[2] * v[2] + v[3] * v[3];
        }
        cbn[t] = nrm;
        __syncthreads();

        // l2-normalize residual
        float s = 0.0f;
#pragma unroll
        for (int i = 0; i < D_LAT; ++i) s += res[i] * res[i];
        float dnm = fmaxf(sqrtf(s), 1e-12f);
        float xin[D_LAT];
#pragma unroll
        for (int i = 0; i < D_LAT; ++i) xin[i] = res[i] / dnm;

        // argmin of ||cb_j||^2 - 2 x.cb_j (row-constant term dropped)
        float best = 3.402823466e+38f;
        int   bi   = 0;
        for (int j = 0; j < N_EMBED; ++j) {
            float dot = 0.0f;
#pragma unroll
            for (int i = 0; i < D_LAT; ++i) dot += xin[i] * cb[j * D_LAT + i];
            float sc = cbn[j] - 2.0f * dot;
            if (sc < best) { best = sc; bi = j; }
        }
        idx[l] = bi;
#pragma unroll
        for (int i = 0; i < D_LAT; ++i) {
            float q    = cb[bi * D_LAT + i];
            float diff = q - xin[i];
            lsum += diff * diff;
            zqt[i] += q;
            res[i] = xin[i] - q;
        }
    }

    // z_q_total as f16 (decoder input)
    _Float16 h[D_LAT];
#pragma unroll
    for (int i = 0; i < D_LAT; ++i) h[i] = (_Float16)zqt[i];
    v8h* o = (v8h*)(zq_out + row * D_LAT);
#pragma unroll
    for (int i = 0; i < 4; ++i) o[i] = ((const v8h*)h)[i];

    // indices (B,3) written as float values
#pragma unroll
    for (int l = 0; l < N_LAYERS; ++l)
        out[IDX_OFF + row * N_LAYERS + l] = (float)idx[l];

    // deterministic block-level loss partial
    red[t] = lsum;
    __syncthreads();
    for (int s2 = 128; s2 > 0; s2 >>= 1) {
        if (t < s2) red[t] += red[t + s2];
        __syncthreads();
    }
    if (t == 0) partial[blockIdx.x] = red[0];
}

__global__ void finalize_loss_kernel(const float* __restrict__ partial,
                                     float* __restrict__ out, int nblk) {
    float s = 0.0f;
    for (int i = 0; i < nblk; ++i) s += partial[i];  // fixed order
    s *= LOSS_SCALE;
    out[LOSS_OFF]   = s;
    out[COMMIT_OFF] = s;
}

// ---------------------------------------------------------------------------
// Launcher
// ---------------------------------------------------------------------------
extern "C" void kernel_launch(void* const* d_in, const int* in_sizes, int n_in,
                              void* d_out, int out_size, void* d_ws,
                              size_t ws_size, hipStream_t stream) {
    (void)in_sizes; (void)n_in; (void)out_size; (void)ws_size;

    const float* x         = (const float*)d_in[0];
    const float* w_f32[8]  = {(const float*)d_in[1], (const float*)d_in[2],
                              (const float*)d_in[3], (const float*)d_in[4],
                              (const float*)d_in[5], (const float*)d_in[6],
                              (const float*)d_in[7], (const float*)d_in[8]};
    const float* codebooks = (const float*)d_in[9];
    float*       out       = (float*)d_out;

    // layer shapes (K, N) in pipeline order: enc1..enc4, dec1..dec4
    const int Ks[8] = {768, 512, 256, 128, 32, 128, 256, 512};
    const int Ns[8] = {512, 256, 128, 32, 128, 256, 512, 768};

    // workspace layout (bytes)
    char*  ws   = (char*)d_ws;
    size_t woff[8];
    size_t off = 0;
    for (int i = 0; i < 8; ++i) {
        woff[i] = off;
        off += (size_t)Ks[i] * Ns[i] * sizeof(_Float16);  // each 256B-multiple
    }
    const size_t PARTIAL_OFF = off;               // 512 floats
    const size_t ACTA_OFF    = 4u * 1024u * 1024u;
    const size_t ACT_BYTES   = (size_t)RQB * 512 * sizeof(_Float16);  // 128 MiB
    const size_t ACTB_OFF    = ACTA_OFF + ACT_BYTES;

    _Float16* actA    = (_Float16*)(ws + ACTA_OFF);
    _Float16* actB    = (_Float16*)(ws + ACTB_OFF);
    float*    zbuf    = (float*)(ws + ACTB_OFF);   // alias actB (dead a2)
    _Float16* zqbuf   = (_Float16*)(ws + ACTA_OFF);// alias actA (dead a3)
    float*    partial = (float*)(ws + PARTIAL_OFF);

    // 1) weights -> f16 fragment-major layout
    for (int i = 0; i < 8; ++i) {
        int elems = Ks[i] * Ns[i];
        convert_w_kernel<<<(elems + 255) / 256, 256, 0, stream>>>(
            w_f32[i], (_Float16*)(ws + woff[i]), Ks[i], Ns[i]);
    }

    const dim3 ggrid(RQB / 128);
    const dim3 gblk(256);

    // 2) encoder
    gemm_wmma_kernel<true, true, false, 8><<<ggrid, gblk, 0, stream>>>(
        (const void*)x, (const _Float16*)(ws + woff[0]), (void*)actA, 768, 512);
    gemm_wmma_kernel<true, false, false, 8><<<ggrid, gblk, 0, stream>>>(
        (const void*)actA, (const _Float16*)(ws + woff[1]), (void*)actB, 512, 256);
    gemm_wmma_kernel<true, false, false, 8><<<ggrid, gblk, 0, stream>>>(
        (const void*)actB, (const _Float16*)(ws + woff[2]), (void*)actA, 256, 128);
    gemm_wmma_kernel<false, false, true, 2><<<ggrid, gblk, 0, stream>>>(
        (const void*)actA, (const _Float16*)(ws + woff[3]), (void*)zbuf, 128, 32);

    // 3) residual VQ (+ indices + loss partials)
    vq_kernel<<<RQB / 256, 256, 0, stream>>>(zbuf, codebooks, zqbuf, out, partial);
    finalize_loss_kernel<<<1, 1, 0, stream>>>(partial, out, RQB / 256);

    // 4) decoder
    gemm_wmma_kernel<true, false, false, 8><<<ggrid, gblk, 0, stream>>>(
        (const void*)zqbuf, (const _Float16*)(ws + woff[4]), (void*)actB, 32, 128);
    gemm_wmma_kernel<true, false, false, 8><<<ggrid, gblk, 0, stream>>>(
        (const void*)actB, (const _Float16*)(ws + woff[5]), (void*)actA, 128, 256);
    gemm_wmma_kernel<true, false, false, 8><<<ggrid, gblk, 0, stream>>>(
        (const void*)actA, (const _Float16*)(ws + woff[6]), (void*)actB, 256, 512);
    gemm_wmma_kernel<false, false, true, 8><<<ggrid, gblk, 0, stream>>>(
        (const void*)actB, (const _Float16*)(ws + woff[7]), (void*)out, 512, 768);
}
